// GlimpseNetwork_29377576304842
// MI455X (gfx1250) — matrix operand order repaired and checked by reading, not compiled, not run on hardware
//
#include <hip/hip_runtime.h>
#include <stdint.h>

#define GSZ   64
#define IMG   512
#define CIN   3
#define NB    64
#define PAD   256            // max_size = GS * 2^(N_GLIMPSES-1)
#define HP    1024           // IMG + 2*PAD
#define CG    9              // N_GLIMPSES * IN_C
#define KDIM  (GSZ*GSZ*CG)   // 36864
#define N1    128
#define N2    256
#define SPLITK 64
#define KC    (KDIM / SPLITK) // 576 (18 k-steps of 32)

typedef __attribute__((ext_vector_type(16))) __bf16 v16bf;
typedef __attribute__((ext_vector_type(8)))  float  v8f;
typedef unsigned int __attribute__((ext_vector_type(4))) u32x4;
typedef int          __attribute__((ext_vector_type(4))) i32x4;
typedef int          __attribute__((ext_vector_type(8))) i32x8;

union Frag32 { u32x4 q[2]; v16bf v; };

__device__ __forceinline__ uint16_t f2bf(float f) {
  uint32_t u = __builtin_bit_cast(uint32_t, f);
  return (uint16_t)((u + 0x7FFFu + ((u >> 16) & 1u)) >> 16);  // RNE truncate
}

__device__ __forceinline__ float keys_cubic(float x) {  // Keys a=-0.5 (jax 'cubic')
  x = fabsf(x);
  if (x <= 1.f) return (1.5f * x - 2.5f) * x * x + 1.f;
  if (x <  2.f) return ((-0.5f * x + 2.5f) * x - 4.f) * x + 2.f;
  return 0.f;
}

// ---- TDM: strided-2D global -> LDS, one DMA per call (issue from ONE wave) ----
// tile_d0 elems/row (2B elems), tile_d1 rows, row stride in elems.
__device__ __forceinline__ void tdm_load_2d_bf16(uint32_t lds_off, const void* gptr,
                                                 uint32_t tile_d0, uint32_t tile_d1,
                                                 uint64_t stride_elems) {
  uint64_t ga = (uint64_t)(uintptr_t)gptr;
  u32x4 g0;
  g0[0] = 1u;                                   // count=1, user descriptor
  g0[1] = lds_off;                              // D# group0[63:32]  lds_addr
  g0[2] = (uint32_t)ga;                         // global_addr[31:0]
  g0[3] = (uint32_t)(ga >> 32) | (2u << 30);    // global_addr[56:32] | type=2
  i32x8 g1;
  g1[0] = (int)(1u << 16);                      // workgroup_mask=0, data_size=1 (2B)
  g1[1] = (int)((tile_d0 & 0xFFFFu) << 16);     // tensor_dim0[15:0] (== tile_d0)
  g1[2] = (int)(((tile_d0 >> 16) & 0xFFFFu) | ((tile_d1 & 0xFFFFu) << 16)); // t_d0 hi | t_d1 lo
  g1[3] = (int)(((tile_d1 >> 16) & 0xFFFFu) | ((tile_d0 & 0xFFFFu) << 16)); // t_d1 hi | tile_dim0
  g1[4] = (int)(tile_d1 & 0xFFFFu);             // tile_dim1 (tile_dim2 = 0)
  g1[5] = (int)(uint32_t)stride_elems;          // tensor_dim0_stride[31:0]
  g1[6] = (int)((stride_elems >> 32) & 0xFFFFu);
  g1[7] = 0;
  i32x4 z4 = (i32x4)0;
#if __has_include(<hip/amd_detail/amd_gfx1250_TDM.h>)
  i32x8 z8 = (i32x8)0;
  __builtin_amdgcn_tensor_load_to_lds(g0, g1, z4, z4, z8, 0);   // clang-23 arity
#else
  __builtin_amdgcn_tensor_load_to_lds(g0, g1, z4, z4, 0);       // ROCm 7.2 arity
#endif
}

// ---- Static antialiased-cubic tap tables (depend only on crop size) ----
__global__ void taps_kernel(int* __restrict__ st1, float* __restrict__ w1,
                            int* __restrict__ st2, float* __restrict__ w2) {
  int o = threadIdx.x;
  if (o >= GSZ) return;
  { // level 1: 128 -> 64, scale 2, 8 taps
    float sf = 2.f, c = (o + 0.5f) * sf - 0.5f;
    int start = (int)floorf(c) - 3;
    float w[8], s = 0.f;
    for (int t = 0; t < 8; ++t) {
      int i = start + t;
      float ww = (i >= 0 && i < 128) ? keys_cubic((i - c) / sf) : 0.f;
      w[t] = ww; s += ww;
    }
    for (int t = 0; t < 8; ++t) w1[o * 8 + t] = w[t] / s;
    st1[o] = start;
  }
  { // level 2: 256 -> 64, scale 4, 16 taps
    float sf = 4.f, c = (o + 0.5f) * sf - 0.5f;
    int start = (int)floorf(c) - 7;
    float w[16], s = 0.f;
    for (int t = 0; t < 16; ++t) {
      int i = start + t;
      float ww = (i >= 0 && i < 256) ? keys_cubic((i - c) / sf) : 0.f;
      w[t] = ww; s += ww;
    }
    for (int t = 0; t < 16; ++t) w2[o * 16 + t] = w[t] / s;
    st2[o] = start;
  }
}

// ---- Wg f32[KDIM][128] -> bf16, packed into WMMA B-fragment layout ----
// B frag (32x16 bf16), this code's k order (same as A): lane holds one column n, 16 contiguous K per lane-half.
// Bp[ (k/16)*128 + n ][ k%16 ]  -> each lane's 16 bf16 are 32 contiguous bytes.
__global__ void pack_wg_kernel(const float* __restrict__ Wg, uint16_t* __restrict__ Bp) {
  size_t idx = (size_t)blockIdx.x * blockDim.x + threadIdx.x;
  if (idx >= (size_t)KDIM * N1) return;
  int n = (int)(idx & (N1 - 1));
  int k = (int)(idx >> 7);
  Bp[(((size_t)(k >> 4) * N1 + n) << 4) + (k & 15)] = f2bf(Wg[idx]);
}

// ---- Glimpse extraction + sparse separable antialiased-cubic resize ----
// Writes A bf16 [NB][KDIM], i = (h*64+w)*9 + level*3 + c  (matches einsum hwc order)
__global__ void glimpse_kernel(const float* __restrict__ img, const float* __restrict__ loc,
                               const int* __restrict__ st1, const float* __restrict__ w1,
                               const int* __restrict__ st2, const float* __restrict__ w2,
                               uint16_t* __restrict__ A) {
  int b = blockIdx.x, level = blockIdx.y;
  int s = GSZ << level;
  float cy = (loc[2 * b + 0] + 1.f) * 0.5f * (float)HP;
  float cx = (loc[2 * b + 1] + 1.f) * 0.5f * (float)HP;
  int tly = (int)rintf(cy - 0.5f * s); tly = tly < 0 ? 0 : (tly > HP - s ? HP - s : tly);
  int tlx = (int)rintf(cx - 0.5f * s); tlx = tlx < 0 ? 0 : (tlx > HP - s ? HP - s : tlx);

  __shared__ float sw[GSZ * 16];
  __shared__ int   sst[GSZ];
  int T = 4 << level;  // taps (levels 1/2)
  if (level > 0) {
    const float* wsrc = (level == 1) ? w1 : w2;
    const int*   ssrc = (level == 1) ? st1 : st2;
    for (int i = threadIdx.x; i < GSZ * T; i += blockDim.x) sw[i] = wsrc[i];
    for (int i = threadIdx.x; i < GSZ;     i += blockDim.x) sst[i] = ssrc[i];
    __syncthreads();
  }

  for (int p = threadIdx.x; p < GSZ * GSZ; p += blockDim.x) {
    int yo = p >> 6, xo = p & 63;
    float v0 = 0.f, v1 = 0.f, v2 = 0.f;
    if (level == 0) {  // 64->64 cubic resize is identity: direct crop copy
      int y = tly - PAD + yo, x = tlx - PAD + xo;
      if (y >= 0 && y < IMG && x >= 0 && x < IMG) {
        size_t base = ((size_t)(b * IMG + y) * IMG + x) * CIN;
        v0 = img[base]; v1 = img[base + 1]; v2 = img[base + 2];
      }
    } else {
      const float* wy = &sw[yo * T]; int sy = sst[yo];
      const float* wx = &sw[xo * T]; int sx = sst[xo];
      for (int ty = 0; ty < T; ++ty) {
        float wyv = wy[ty];
        if (wyv == 0.f) continue;
        int y = tly + sy + ty - PAD;
        if (y < 0 || y >= IMG) continue;        // zero-padded region
        float r0 = 0.f, r1 = 0.f, r2 = 0.f;
        size_t rowb = (size_t)(b * IMG + y) * IMG;
        for (int tx = 0; tx < T; ++tx) {
          float wxv = wx[tx];
          int x = tlx + sx + tx - PAD;
          if (wxv != 0.f && x >= 0 && x < IMG) {
            size_t base = (rowb + x) * CIN;
            r0 += wxv * img[base]; r1 += wxv * img[base + 1]; r2 += wxv * img[base + 2];
          }
        }
        v0 += wyv * r0; v1 += wyv * r1; v2 += wyv * r2;
      }
    }
    size_t o = (size_t)b * KDIM + (size_t)p * CG + level * CIN;
    A[o] = f2bf(v0); A[o + 1] = f2bf(v1); A[o + 2] = f2bf(v2);
  }
}

// ---- Split-K bf16 WMMA GEMM with TDM-staged A in LDS ----
// Per WG: TDM pulls the whole A chunk (64 x 576 bf16 = 72 KB, row stride KDIM) into
// LDS once (A is reused by all 8 waves); B streams from global non-temporally and is
// double-buffered in registers so its latency hides under the WMMAs.
__global__ void __launch_bounds__(256, 2)
gemm_kernel(const uint16_t* __restrict__ A, const uint16_t* __restrict__ Bp,
            float* __restrict__ partial) {
  extern __shared__ uint16_t ldsA[];            // NB*KC bf16 = 73728 bytes
  int s    = blockIdx.x;
  int wave = threadIdx.x >> 5;
  int lane = threadIdx.x & 31;
  int nl = lane & 15, kh = lane >> 4;
  int ncol = wave * 16 + nl;
  int k0 = s * KC;

  if (wave == 0) {                              // one DMA per workgroup
    tdm_load_2d_bf16((uint32_t)(uintptr_t)ldsA, A + k0, KC, NB, KDIM);
    __builtin_amdgcn_s_wait_tensorcnt(0);
  }
  __syncthreads();

  // per-lane base for B fragments (32 contiguous bytes per (k-half, column) group)
  const uint16_t* bbase = Bp + ((((size_t)(k0 >> 4) + kh) * N1 + ncol) << 4);

  auto loadB = [&](int kk) {
    Frag32 f;
    const u32x4* bp = (const u32x4*)(bbase + (((size_t)(kk >> 4)) * N1 << 4));
    f.q[0] = __builtin_nontemporal_load(bp);
    f.q[1] = __builtin_nontemporal_load(bp + 1);
    return f;
  };

  v8f acc[4] = {};
  Frag32 bf = loadB(0);
  for (int kk = 0; kk < KC; kk += 32) {
    // prefetch B ~8 k-steps ahead (global_prefetch_b8)
    __builtin_prefetch(bbase + ((((size_t)(kk >> 4)) + 16) * N1 << 4), 0, 1);
    // register double-buffer: issue next B load before this step's WMMAs
    Frag32 bnext = (kk + 32 < KC) ? loadB(kk + 32) : bf;
    // batch all 4 A fragments (8 ds_load_b128) so dscnt waits stagger
    Frag32 af[4];
#pragma unroll
    for (int mt = 0; mt < 4; ++mt) {
      const uint16_t* ap = ldsA + (mt * 16 + nl) * KC + kk + kh * 16;
      af[mt].q[0] = *(const u32x4*)ap;         // K (kh*16)..(kh*16+7)
      af[mt].q[1] = *(const u32x4*)(ap + 8);   // K (kh*16+8)..(kh*16+15)
    }
#pragma unroll
    for (int mt = 0; mt < 4; ++mt)
      acc[mt] = __builtin_amdgcn_wmma_f32_16x16x32_bf16(
          false, af[mt].v, false, bf.v, (short)0, acc[mt], false, false);
    bf = bnext;
  }
#pragma unroll
  for (int mt = 0; mt < 4; ++mt)
#pragma unroll
    for (int r = 0; r < 8; ++r) {
      int m = mt * 16 + r + 8 * kh;   // C/D layout: VGPR r -> M=r (+8 for hi lanes)
      __builtin_nontemporal_store(acc[mt][r],
          partial + ((size_t)s * NB + m) * N1 + ncol);
    }
}

// ---- Deterministic split-K reduce + bias/relu + loc-MLP + 128->256 output ----
__global__ void epilogue_kernel(const float* __restrict__ partial, const float* __restrict__ loc,
                                const float* __restrict__ bg, const float* __restrict__ Wl,
                                const float* __restrict__ bl, const float* __restrict__ Wo,
                                const float* __restrict__ bo, float* __restrict__ out) {
  int b = blockIdx.x, tid = threadIdx.x;
  __shared__ float xs[N1];
  if (tid < N1) {
    float a = 0.f;
    for (int s = 0; s < SPLITK; ++s) a += partial[((size_t)s * NB + b) * N1 + tid];
    float x1 = fmaxf(a + bg[tid], 0.f);
    float x2 = fmaxf(loc[2 * b] * Wl[tid] + loc[2 * b + 1] * Wl[N1 + tid] + bl[tid], 0.f);
    xs[tid] = x1 + x2;
  }
  __syncthreads();
  float o = bo[tid];
  for (int k = 0; k < N1; ++k) o += xs[k] * Wo[k * N2 + tid];
  out[(size_t)b * N2 + tid] = fmaxf(o, 0.f);
}

extern "C" void kernel_launch(void* const* d_in, const int* in_sizes, int n_in,
                              void* d_out, int out_size, void* d_ws, size_t ws_size,
                              hipStream_t stream) {
  const float* image = (const float*)d_in[0];
  const float* loc   = (const float*)d_in[1];
  const float* Wg    = (const float*)d_in[2];
  const float* bg    = (const float*)d_in[3];
  const float* Wl    = (const float*)d_in[4];
  const float* bl    = (const float*)d_in[5];
  const float* Wo    = (const float*)d_in[6];
  const float* bo    = (const float*)d_in[7];
  float* out = (float*)d_out;

  char* ws = (char*)d_ws;
  size_t off = 0;
  auto alloc = [&](size_t bytes) -> void* {
    void* p = ws + off;
    off = (off + bytes + 255) & ~(size_t)255;
    return p;
  };
  uint16_t* A   = (uint16_t*)alloc((size_t)NB * KDIM * 2);        // 4.72 MB bf16 glimpses
  uint16_t* Bp  = (uint16_t*)alloc((size_t)KDIM * N1 * 2);        // 9.44 MB packed bf16 Wg
  float*    par = (float*)   alloc((size_t)SPLITK * NB * N1 * 4); // 2 MB split-K partials
  int*   st1 = (int*)  alloc(GSZ * 4);
  int*   st2 = (int*)  alloc(GSZ * 4);
  float* w1  = (float*)alloc(GSZ * 8 * 4);
  float* w2  = (float*)alloc(GSZ * 16 * 4);

  taps_kernel<<<1, 64, 0, stream>>>(st1, w1, st2, w2);
  pack_wg_kernel<<<(KDIM * N1) / 256, 256, 0, stream>>>(Wg, Bp);
  glimpse_kernel<<<dim3(NB, 3), 256, 0, stream>>>(image, loc, st1, w1, st2, w2, A);
  gemm_kernel<<<SPLITK, 256, NB * KC * 2, stream>>>(A, Bp, par);
  epilogue_kernel<<<NB, 256, 0, stream>>>(par, loc, bg, Wl, bl, Wo, bo, out);
}
